// FiLMedConvBlock_context_81226421502655
// MI455X (gfx1250) — matrix operand order repaired and checked by reading, not compiled, not run on hardware
//
#include <hip/hip_runtime.h>

// ---------------- problem constants ----------------
#define BB   64
#define DD   512
#define HW   1024          // H*W
#define NN   20
#define EE   512
#define CIN  520           // D + 8 coord channels
#define KPAD 544           // CIN padded to multiple of 32 (17 * 32)
#define KSTEPS 17

typedef __attribute__((ext_vector_type(16))) __bf16 v16bf;
typedef __attribute__((ext_vector_type(8)))  float  v8f;

union BFrag { v16bf v; unsigned int u[8]; unsigned short s[16]; };

__device__ __forceinline__ unsigned short f2bf(float f) {
    union { float f; unsigned int u; } x; x.f = f;
    unsigned int u = x.u;
    u += 0x7FFFu + ((u >> 16) & 1u);   // round-to-nearest-even
    return (unsigned short)(u >> 16);
}

__device__ __forceinline__ unsigned int pack2bf(float lo, float hi) {
    return (unsigned int)f2bf(lo) | ((unsigned int)f2bf(hi) << 16);
}

// A-fragment k offset for 16-bit A 16x32, contiguous order (k = 16*hi + s):
// VGPR v, lane-half hi -> k pair base
__device__ __forceinline__ int a_kbase(int v, int hi) {
    return (hi << 4) + (v << 1);
}

// ---------------- prep: weights to bf16 ----------------
__global__ __launch_bounds__(256) void prep_wb_kernel(
    const float* __restrict__ w, unsigned short* __restrict__ wb) {
    int idx = blockIdx.x * 256 + threadIdx.x;
    if (idx >= 512 * KPAD) return;
    int o = idx / KPAD, k = idx - o * KPAD;
    float v = (k < CIN) ? w[o * CIN + k] : 0.0f;
    wb[idx] = f2bf(v);
}

__global__ __launch_bounds__(256) void prep_gwb_kernel(
    const float* __restrict__ w, unsigned short* __restrict__ wb) {
    int idx = blockIdx.x * 256 + threadIdx.x;
    if (idx < 2 * EE * DD) wb[idx] = f2bf(w[idx]);
}

// ---------------- K1: tile_visu = mean over HW ----------------
__global__ __launch_bounds__(256) void tilevisu_kernel(
    const float* __restrict__ fv, float* __restrict__ tv) {
    __shared__ float r[256];
    const float4* p = (const float4*)(fv + (size_t)blockIdx.x * HW);
    float4 v = p[threadIdx.x];                 // 1024 = 256 threads * 4
    r[threadIdx.x] = (v.x + v.y) + (v.z + v.w);
    __syncthreads();
    for (int off = 128; off > 0; off >>= 1) {
        if (threadIdx.x < off) r[threadIdx.x] += r[threadIdx.x + off];
        __syncthreads();
    }
    if (threadIdx.x == 0) tv[blockIdx.x] = r[0] * (1.0f / HW);
}

// ---------------- K2: attention + masked softmax + attn_lang ----------------
// one block per batch, 256 threads
__global__ __launch_bounds__(256) void attn_kernel(
    const float* __restrict__ fword, const float* __restrict__ cs,
    const int*   __restrict__ wmask, const float* __restrict__ tv,
    const float* __restrict__ w1, const float* __restrict__ b1,
    const float* __restrict__ w2, const float* __restrict__ b2,
    float* __restrict__ out_score, float* __restrict__ out_alang,
    unsigned short* __restrict__ alang_bf) {
    extern __shared__ char smem[];
    float* p   = (float*)smem;          // [NN][DD] : tv*fword*cs
    float* af  = p  + NN * DD;          // [256][NN]
    float* sc  = af + 256 * NN;         // [NN]
    float* ms  = sc + NN;               // [NN]
    int*   imk = (int*)(ms + NN);       // [NN]

    const int b = blockIdx.x, tid = threadIdx.x;
    const float* fw = fword + (size_t)b * NN * DD;

    for (int idx = tid; idx < NN * DD; idx += 256) {
        int n = idx / DD, d = idx - n * DD;
        p[idx] = fw[idx] * tv[b * DD + d] * cs[b * NN + n];
    }
    __syncthreads();

    {   // attn_feat row per thread: af[o][n] = tanh(b1[o] + sum_d w1[o][d]*p[n][d])
        const int o = tid;
        float a[NN];
        #pragma unroll
        for (int n = 0; n < NN; ++n) a[n] = b1[o];
        const float* wr = w1 + (size_t)o * DD;
        for (int d = 0; d < DD; d += 4) {
            float4 w = *(const float4*)(wr + d);
            #pragma unroll
            for (int n = 0; n < NN; ++n) {
                float4 pv = *(const float4*)(p + n * DD + d);   // ds_load_b128 broadcast
                a[n] += w.x * pv.x + w.y * pv.y + w.z * pv.z + w.w * pv.w;
            }
        }
        #pragma unroll
        for (int n = 0; n < NN; ++n) af[o * NN + n] = tanhf(a[n]);
    }
    __syncthreads();

    if (tid < NN) {
        float s = b2[0];
        for (int o = 0; o < 256; ++o) s += w2[o] * af[o * NN + tid];
        sc[tid] = s;
        out_score[b * NN + tid] = s;
        imk[tid] = wmask[b * NN + tid];
    }
    __syncthreads();

    if (tid == 0) {
        float mx = -1e30f;
        #pragma unroll
        for (int n = 0; n < NN; ++n) mx = fmaxf(mx, sc[n] * 10.0f);
        float e[NN], sum = 0.0f;
        #pragma unroll
        for (int n = 0; n < NN; ++n) { e[n] = __expf(sc[n] * 10.0f - mx); sum += e[n]; }
        imk[0] = 0;
        int idx = 0;
        #pragma unroll
        for (int n = 0; n < NN; ++n) idx += imk[n];
        imk[idx] = 0;                        // idx <= 19 since imk[0]==0
        float den = 1e-8f, msv[NN];
        #pragma unroll
        for (int n = 0; n < NN; ++n) { msv[n] = (e[n] / sum) * (float)imk[n]; den += msv[n]; }
        float rden = 1.0f / den;
        #pragma unroll
        for (int n = 0; n < NN; ++n) ms[n] = msv[n] * rden;
    }
    __syncthreads();

    for (int d = tid; d < DD; d += 256) {
        float al = 0.0f;
        #pragma unroll
        for (int n = 0; n < NN; ++n) al += ms[n] * fw[n * DD + d];
        out_alang[b * DD + d] = al;
        alang_bf[b * DD + d]  = f2bf(al);
    }
}

// ---------------- K3: film = attn_lang @ gamma_w^T, WMMA bf16 ----------------
// M=64 (4 tiles of batch), N=1024 (64 tiles), K=512 (16 steps)
// grid = 32 blocks x 8 waves = 256 waves, one 16x16 tile each
__global__ __launch_bounds__(256) void film_kernel(
    const unsigned short* __restrict__ alb,   // [64][512] bf16
    const unsigned short* __restrict__ gwb,   // [1024][512] bf16
    const float* __restrict__ gb,             // [1024]
    float* __restrict__ gamma_ws, float* __restrict__ beta_ws) {
    const int tid = threadIdx.x, lane = tid & 31, wid = tid >> 5;
    const int wg = blockIdx.x * 8 + wid;
    const int mt = wg >> 6, nt = wg & 63;
    const int m = lane & 15, hi = lane >> 4;

    v8f acc = 0.0f;
    for (int kt = 0; kt < 16; ++kt) {
        BFrag a, bf;
        #pragma unroll
        for (int v = 0; v < 8; ++v) {
            int k = kt * 32 + a_kbase(v, hi);
            a.u[v] = *(const unsigned int*)(alb + (size_t)(mt * 16 + m) * DD + k);
        }
        const uint4* bp = (const uint4*)(gwb + (size_t)(nt * 16 + m) * DD + kt * 32 + hi * 16);
        uint4 q0 = bp[0], q1 = bp[1];
        bf.u[0] = q0.x; bf.u[1] = q0.y; bf.u[2] = q0.z; bf.u[3] = q0.w;
        bf.u[4] = q1.x; bf.u[5] = q1.y; bf.u[6] = q1.z; bf.u[7] = q1.w;
        acc = __builtin_amdgcn_wmma_f32_16x16x32_bf16(false, a.v, false, bf.v,
                                                      (short)0, acc, false, false);
    }
    #pragma unroll
    for (int g = 0; g < 8; ++g) {
        int batch = mt * 16 + g + 8 * hi;
        int n = nt * 16 + m;
        float val = tanhf(acc[g] + gb[n]);
        if (n < EE) gamma_ws[batch * EE + n] = val;
        else        beta_ws [batch * EE + (n - EE)] = val;
    }
}

// ---------------- K4: fused conv GEMM + InstanceNorm + FiLM + ReLU + residual ----------------
// grid = (32 o-tiles, 64 batches), 256 threads = 8 waves
// block computes 16 output channels x all 1024 pixels; K = 520 (padded 544)
// LDS: x panel bf16, column-major [1024 cols][stride 40 halves] = 81920 B
__global__ __launch_bounds__(256) void conv_norm_kernel(
    const float* __restrict__ fvisu, const float* __restrict__ fcoord,
    const unsigned short* __restrict__ wb, const float* __restrict__ cb,
    const float* __restrict__ gamma_ws, const float* __restrict__ beta_ws,
    float* __restrict__ out) {
    extern __shared__ char smem[];
    unsigned short* xl = (unsigned short*)smem;   // [1024 cols][stride 40] halves

    const int b  = blockIdx.y;
    const int o0 = blockIdx.x * 16;
    const int tid = threadIdx.x;
    const int lane = tid & 31, wid = tid >> 5;
    const int m = lane & 15, hi = lane >> 4;

    const float* xv = fvisu  + (size_t)b * DD * HW;
    const float* xc = fcoord + (size_t)b * 8  * HW;

    v8f acc[8];
    #pragma unroll
    for (int t = 0; t < 8; ++t) acc[t] = 0.0f;

    const int col4 = (tid & 255) * 4;             // 4 consecutive pixel columns per thread

    for (int kt = 0; kt < KSTEPS; ++kt) {
        // stage 32-channel x panel into LDS as bf16, column-major.
        // per iteration: channel pair (c0,c0+1) x 4 columns:
        //   2x global_load_b128 + 4x packed ds_store_b32
        for (int cp = 0; cp < 16; ++cp) {
            int c0 = cp * 2;
            int cg = kt * 32 + c0;
            float4 r0, r1;
            if (cg < DD) {
                r0 = *(const float4*)(xv + (size_t)cg * HW + col4);
                r1 = *(const float4*)(xv + (size_t)(cg + 1) * HW + col4);
            } else if (cg < CIN) {
                r0 = *(const float4*)(xc + (size_t)(cg - DD) * HW + col4);
                r1 = *(const float4*)(xc + (size_t)(cg + 1 - DD) * HW + col4);
            } else {
                r0 = make_float4(0.f, 0.f, 0.f, 0.f);
                r1 = r0;
            }
            unsigned int* dst0 = (unsigned int*)(xl + (col4 + 0) * 40 + c0);
            unsigned int* dst1 = (unsigned int*)(xl + (col4 + 1) * 40 + c0);
            unsigned int* dst2 = (unsigned int*)(xl + (col4 + 2) * 40 + c0);
            unsigned int* dst3 = (unsigned int*)(xl + (col4 + 3) * 40 + c0);
            *dst0 = pack2bf(r0.x, r1.x);
            *dst1 = pack2bf(r0.y, r1.y);
            *dst2 = pack2bf(r0.z, r1.z);
            *dst3 = pack2bf(r0.w, r1.w);
        }
        __syncthreads();

        // A fragment: weights wb[o0+m][kt*32 + kmap] (global, L2-resident)
        BFrag a;
        #pragma unroll
        for (int v = 0; v < 8; ++v) {
            int k = kt * 32 + a_kbase(v, hi);
            a.u[v] = *(const unsigned int*)(wb + (size_t)(o0 + m) * KPAD + k);
        }

        #pragma unroll
        for (int t = 0; t < 8; ++t) {
            int col = wid * 128 + t * 16 + m;             // pixel column for this lane
            const uint4* bp = (const uint4*)(xl + col * 40 + hi * 16);
            uint4 q0 = bp[0], q1 = bp[1];                 // ds_load_b128 x2
            BFrag bf;
            bf.u[0] = q0.x; bf.u[1] = q0.y; bf.u[2] = q0.z; bf.u[3] = q0.w;
            bf.u[4] = q1.x; bf.u[5] = q1.y; bf.u[6] = q1.z; bf.u[7] = q1.w;
            acc[t] = __builtin_amdgcn_wmma_f32_16x16x32_bf16(false, a.v, false, bf.v,
                                                             (short)0, acc[t], false, false);
        }
        __syncthreads();
    }

    // bias
    #pragma unroll
    for (int t = 0; t < 8; ++t) {
        #pragma unroll
        for (int g = 0; g < 8; ++g) acc[t][g] += cb[o0 + g + 8 * hi];
    }

    // InstanceNorm stats over the full 1024 pixels per output row (LDS reduce)
    float* red = (float*)smem;     // [16 rows][2] : sum, sumsq (panel no longer needed)
    if (tid < 32) red[tid] = 0.0f;
    __syncthreads();
    #pragma unroll
    for (int g = 0; g < 8; ++g) {
        float s = 0.0f, s2 = 0.0f;
        #pragma unroll
        for (int t = 0; t < 8; ++t) { float v = acc[t][g]; s += v; s2 += v * v; }
        int row = g + 8 * hi;
        atomicAdd(&red[row * 2 + 0], s);
        atomicAdd(&red[row * 2 + 1], s2);
    }
    __syncthreads();

    const float inv = 1.0f / (float)HW;
    float gmu[8], grs[8], gga[8], gbe[8];
    #pragma unroll
    for (int g = 0; g < 8; ++g) {
        int row = g + 8 * hi;
        int o = o0 + row;
        float mu  = red[row * 2 + 0] * inv;
        float var = red[row * 2 + 1] * inv - mu * mu;
        gmu[g] = mu;
        grs[g] = rsqrtf(var + 1e-5f);
        gga[g] = gamma_ws[b * EE + o];
        gbe[g] = beta_ws [b * EE + o];
    }

    #pragma unroll
    for (int t = 0; t < 8; ++t) {
        int col = wid * 128 + t * 16 + m;
        #pragma unroll
        for (int g = 0; g < 8; ++g) {
            int o = o0 + g + 8 * hi;
            size_t idx = ((size_t)b * DD + o) * HW + col;
            float nv = (acc[t][g] - gmu[g]) * grs[g];
            float y  = gga[g] * nv + gbe[g];
            y = y > 0.0f ? y : 0.0f;
            // write-once output: non-temporal keeps L2 for shared x panels
            __builtin_nontemporal_store(fvisu[idx] + y, &out[idx]);
        }
    }
}

// ---------------- launch ----------------
extern "C" void kernel_launch(void* const* d_in, const int* in_sizes, int n_in,
                              void* d_out, int out_size, void* d_ws, size_t ws_size,
                              hipStream_t stream) {
    const float* fvisu   = (const float*)d_in[0];
    const float* fword   = (const float*)d_in[1];
    const float* cscore  = (const float*)d_in[2];
    const float* fcoord  = (const float*)d_in[3];
    // d_in[4] = gest : unused by the reference
    const int*   wmask   = (const int*)  d_in[5];
    const float* w1      = (const float*)d_in[6];
    const float* b1      = (const float*)d_in[7];
    const float* w2      = (const float*)d_in[8];
    const float* b2      = (const float*)d_in[9];
    const float* gamma_w = (const float*)d_in[10];
    const float* gamma_b = (const float*)d_in[11];
    const float* conv1_w = (const float*)d_in[12];
    const float* conv1_b = (const float*)d_in[13];

    float* out = (float*)d_out;
    float* out_fmodu = out;                                   // 64*512*1024
    float* out_alang = out + (size_t)BB * DD * HW;            // 64*512
    float* out_score = out_alang + (size_t)BB * DD;           // 64*20

    // workspace layout (bytes)
    char* ws = (char*)d_ws;
    float*          tv        = (float*)(ws + 0);             // 131072 B
    float*          gamma_ws  = (float*)(ws + 131072);        // 131072 B
    float*          beta_ws   = (float*)(ws + 262144);        // 131072 B
    unsigned short* alang_bf  = (unsigned short*)(ws + 393216);   //  65536 B
    unsigned short* wb        = (unsigned short*)(ws + 458752);   // 557056 B
    unsigned short* gwb       = (unsigned short*)(ws + 1015808);  // 1048576 B

    // weight conversion (independent of data kernels)
    prep_wb_kernel <<<dim3((512 * KPAD + 255) / 256), dim3(256), 0, stream>>>(conv1_w, wb);
    prep_gwb_kernel<<<dim3((2 * EE * DD + 255) / 256), dim3(256), 0, stream>>>(gamma_w, gwb);

    // K1: tile_visu
    tilevisu_kernel<<<dim3(BB * DD), dim3(256), 0, stream>>>(fvisu, tv);

    // K2: attention pipeline (per batch)
    size_t k2_lds = (size_t)(NN * DD + 256 * NN + NN + NN) * 4 + NN * 4;
    attn_kernel<<<dim3(BB), dim3(256), k2_lds, stream>>>(
        fword, cscore, wmask, tv, w1, b1, w2, b2,
        out_score, out_alang, alang_bf);

    // K3: film GEMM (WMMA bf16) -> gamma/beta
    film_kernel<<<dim3(32), dim3(256), 0, stream>>>(alang_bf, gwb, gamma_b, gamma_ws, beta_ws);

    // K4: fused conv GEMM + InstanceNorm + FiLM + ReLU + residual (WMMA bf16)
    const int k4_lds = 1024 * 40 * 2;  // 81920 B
    hipFuncSetAttribute((const void*)conv_norm_kernel,
                        hipFuncAttributeMaxDynamicSharedMemorySize, k4_lds);
    conv_norm_kernel<<<dim3(32, BB), dim3(256), k4_lds, stream>>>(
        fvisu, fcoord, wb, conv1_b, gamma_ws, beta_ws, out_fmodu);
}
